// BinaryGroupConv_83803401879818
// MI455X (gfx1250) — 8-bit matrix operand order repaired and checked by reading, compiled, not run on hardware
//
#include <hip/hip_runtime.h>


typedef __attribute__((ext_vector_type(8))) int v8i;

#define BATCH 32
#define CHN   256
#define HIN   56
#define HOUT  28
#define PIX   (BATCH * HOUT * HOUT)
#define EPSBN 1e-5f

__device__ __forceinline__ signed char sign8(float v) {
    return (signed char)((v > 0.0f) ? 1 : ((v < 0.0f) ? -1 : 0));
}

__global__ __launch_bounds__(256) void binarize_x_kernel(
    const float* __restrict__ x, signed char* __restrict__ xbin) {
    __shared__ signed char tile[HIN * 260];
    const int bh = blockIdx.x;
    const int b = bh / HIN;
    const int h = bh % HIN;
    const int tid = threadIdx.x;
    #pragma unroll 4
    for (int i = 0; i < 56; ++i) {
        int e = i * 256 + tid;
        int c = e / HIN;
        int w = e % HIN;
        float v = x[(((size_t)b * CHN + c) * HIN + h) * HIN + w];
        tile[w * 260 + c] = sign8(v);
    }
    __syncthreads();
    const int* src = (const int*)tile;
    int* dst = (int*)(xbin + (size_t)bh * HIN * CHN);
    #pragma unroll
    for (int i = 0; i < 14; ++i) {
        int d = i * 256 + tid;
        int w = d >> 6;
        int dw = d & 63;
        dst[d] = src[w * 65 + dw];
    }
}

__global__ __launch_bounds__(256) void maxpool_kernel(
    const float* __restrict__ x, float* __restrict__ shortcut) {
    __shared__ float tile[HOUT * 257];
    const int bo = blockIdx.x;
    const int b = bo / HOUT;
    const int oh = bo % HOUT;
    const int tid = threadIdx.x;
    #pragma unroll 4
    for (int i = 0; i < 28; ++i) {
        int e = i * 256 + tid;
        int c = e / HOUT;
        int ow = e % HOUT;
        float m = -__builtin_huge_valf();
        #pragma unroll
        for (int kh = 0; kh < 3; ++kh) {
            int ih = 2 * oh - 1 + kh;
            if (ih < 0 || ih >= HIN) continue;
            #pragma unroll
            for (int kw = 0; kw < 3; ++kw) {
                int iw = 2 * ow - 1 + kw;
                if (iw < 0 || iw >= HIN) continue;
                m = fmaxf(m, x[(((size_t)b * CHN + c) * HIN + ih) * HIN + iw]);
            }
        }
        tile[ow * 257 + c] = m;
    }
    __syncthreads();
    float* dst = shortcut + (size_t)bo * HOUT * CHN;
    #pragma unroll
    for (int i = 0; i < 28; ++i) {
        int d = i * 256 + tid;
        int ow = d >> 8;
        int c = d & 255;
        dst[d] = tile[ow * 257 + c];
    }
}

__global__ void pack_w1_kernel(const float* __restrict__ w1,
                               signed char* __restrict__ wp) {
    int tid = blockIdx.x * blockDim.x + threadIdx.x;
    int lane = tid & 31;
    int f = tid >> 5;
    int j = f % 9;
    int t = (f / 9) % 4;
    int g = f / 36;
    int kh = j / 3, kw = j % 3;
    int n = lane & 15;
    int half = lane >> 4;
    int oc = g * 64 + t * 16 + n;
    int* dst = (int*)(wp + (size_t)f * 1024 + lane * 32);
    #pragma unroll
    for (int v = 0; v < 8; ++v) {
        unsigned int word = 0;
        #pragma unroll
        for (int bb = 0; bb < 4; ++bb) {
            int kk = (v >> 1) * 16 + half * 8 + (v & 1) * 4 + bb;
            float wv = w1[(((size_t)oc * 64 + kk) * 3 + kh) * 3 + kw];
            word |= ((unsigned int)(unsigned char)sign8(wv)) << (8 * bb);
        }
        dst[v] = (int)word;
    }
}

__global__ void pack_w2_kernel(const float* __restrict__ w2,
                               signed char* __restrict__ wp) {
    int tid = blockIdx.x * blockDim.x + threadIdx.x;
    int lane = tid & 31;
    int f = tid >> 5;
    int j = f & 3;
    int t = f >> 2;
    int n = lane & 15;
    int half = lane >> 4;
    int oc = t * 16 + n;
    int* dst = (int*)(wp + (size_t)f * 1024 + lane * 32);
    #pragma unroll
    for (int v = 0; v < 8; ++v) {
        unsigned int word = 0;
        #pragma unroll
        for (int bb = 0; bb < 4; ++bb) {
            int kk = (v >> 1) * 16 + half * 8 + (v & 1) * 4 + bb;
            int ic = j * 64 + kk;
            float wv = w2[(size_t)oc * CHN + ic];
            word |= ((unsigned int)(unsigned char)sign8(wv)) << (8 * bb);
        }
        dst[v] = (int)word;
    }
}

__global__ void zero_stats_kernel(float* __restrict__ s) {
    s[blockIdx.x * blockDim.x + threadIdx.x] = 0.0f;
}

__global__ __launch_bounds__(128) void conv1_wmma_kernel(
    const signed char* __restrict__ xbin,
    const signed char* __restrict__ wpack,
    float* __restrict__ ybuf,
    float* __restrict__ gsum,
    float* __restrict__ gsumsq)
{
    __shared__ float ssum[64];
    __shared__ float ssq[64];
    if (threadIdx.x < 64) { ssum[threadIdx.x] = 0.0f; ssq[threadIdx.x] = 0.0f; }
    __syncthreads();

    const int wave = threadIdx.x >> 5;
    const int lane = threadIdx.x & 31;
    const int mt = blockIdx.x >> 2;
    const int g  = blockIdx.x & 3;
    const int t  = wave;

    const int r    = lane & 15;
    const int half = lane >> 4;
    const int p    = mt * 16 + r;
    const int b    = p / (HOUT * HOUT);
    const int rem  = p % (HOUT * HOUT);
    const int oh   = rem / HOUT;
    const int ow   = rem % HOUT;

    v8i acc = {};
    const signed char* wbase =
        wpack + (size_t)((g * 4 + t) * 9) * 1024 + lane * 32;

    #pragma unroll
    for (int j = 0; j < 9; ++j) {
        const int kh = j / 3, kw = j % 3;
        const int ih = 2 * oh - 1 + kh;
        const int iw = 2 * ow - 1 + kw;
        const bool valid = (ih >= 0) & (ih < HIN) & (iw >= 0) & (iw < HIN);

        v8i a = {};
        if (valid) {
            const int* ap = (const int*)(xbin +
                (((size_t)(b * HIN + ih) * HIN + iw) << 8) + g * 64 + half * 8);
            a[0] = ap[0];  a[1] = ap[1];
            a[2] = ap[4];  a[3] = ap[5];
            a[4] = ap[8];  a[5] = ap[9];
            a[6] = ap[12]; a[7] = ap[13];
        }
        v8i bf;
        const int* bp = (const int*)(wbase + (size_t)j * 1024);
        #pragma unroll
        for (int q = 0; q < 8; ++q) bf[q] = bp[q];
        if (j < 8) __builtin_prefetch(wbase + (size_t)(j + 1) * 1024, 0, 1);

        acc = __builtin_amdgcn_wmma_i32_16x16x64_iu8(
                  true, a, true, bf, acc, false, false);
    }

    const int n = lane & 15;
    const int cbase = g * 64 + t * 16 + n;
    float s = 0.0f, s2 = 0.0f;
    #pragma unroll
    for (int v = 0; v < 8; ++v) {
        const int m = v + 8 * half;
        float fv = (float)acc[v];
        s += fv;
        s2 += fv * fv;
        ybuf[(size_t)(mt * 16 + m) * CHN + cbase] = fv;
    }
    const int lc = t * 16 + n;
    atomicAdd(&ssum[lc], s);
    atomicAdd(&ssq[lc], s2);
    __syncthreads();
    if (threadIdx.x < 64) {
        atomicAdd(&gsum[g * 64 + threadIdx.x], ssum[threadIdx.x]);
        atomicAdd(&gsumsq[g * 64 + threadIdx.x], ssq[threadIdx.x]);
    }
}

__global__ void bn1_apply_kernel(const float* __restrict__ conv,
                                 const float* __restrict__ shortcut,
                                 const float* __restrict__ sum,
                                 const float* __restrict__ sumsq,
                                 const float* __restrict__ gamma,
                                 const float* __restrict__ beta,
                                 float* __restrict__ x1,
                                 signed char* __restrict__ x1bin) {
    int idx = blockIdx.x * blockDim.x + threadIdx.x;
    int c = idx & (CHN - 1);
    const float inv = 1.0f / (float)PIX;
    float mean = sum[c] * inv;
    float var = sumsq[c] * inv - mean * mean;
    float rs = rsqrtf(var + EPSBN);
    float y = (conv[idx] - mean) * rs * gamma[c] + beta[c] + shortcut[idx];
    x1[idx] = y;
    x1bin[idx] = sign8(y);
}

__global__ __launch_bounds__(128) void conv2_wmma_kernel(
    const signed char* __restrict__ xbin,
    const signed char* __restrict__ wpack,
    float* __restrict__ ybuf,
    float* __restrict__ gsum,
    float* __restrict__ gsumsq)
{
    __shared__ float ssum[64];
    __shared__ float ssq[64];
    if (threadIdx.x < 64) { ssum[threadIdx.x] = 0.0f; ssq[threadIdx.x] = 0.0f; }
    __syncthreads();

    const int wave = threadIdx.x >> 5;
    const int lane = threadIdx.x & 31;
    const int mt = blockIdx.x >> 2;
    const int nt = (blockIdx.x & 3) * 4 + wave;

    const int r    = lane & 15;
    const int half = lane >> 4;
    const int p    = mt * 16 + r;

    const signed char* arow = xbin + (size_t)p * CHN + half * 8;
    const signed char* wbase = wpack + (size_t)(nt * 4) * 1024 + lane * 32;

    v8i acc = {};
    #pragma unroll
    for (int j = 0; j < 4; ++j) {
        const int* ap = (const int*)(arow + j * 64);
        v8i a;
        a[0] = ap[0];  a[1] = ap[1];
        a[2] = ap[4];  a[3] = ap[5];
        a[4] = ap[8];  a[5] = ap[9];
        a[6] = ap[12]; a[7] = ap[13];

        v8i bf;
        const int* bp = (const int*)(wbase + (size_t)j * 1024);
        #pragma unroll
        for (int q = 0; q < 8; ++q) bf[q] = bp[q];
        if (j < 3) __builtin_prefetch(wbase + (size_t)(j + 1) * 1024, 0, 1);

        acc = __builtin_amdgcn_wmma_i32_16x16x64_iu8(
                  true, a, true, bf, acc, false, false);
    }

    const int n = lane & 15;
    const int cbase = nt * 16 + n;
    float s = 0.0f, s2 = 0.0f;
    #pragma unroll
    for (int v = 0; v < 8; ++v) {
        const int m = v + 8 * half;
        float fv = (float)acc[v];
        s += fv;
        s2 += fv * fv;
        ybuf[(size_t)(mt * 16 + m) * CHN + cbase] = fv;
    }
    const int lc = wave * 16 + n;
    atomicAdd(&ssum[lc], s);
    atomicAdd(&ssq[lc], s2);
    __syncthreads();
    if (threadIdx.x < 64) {
        int c = (blockIdx.x & 3) * 64 + threadIdx.x;
        atomicAdd(&gsum[c], ssum[threadIdx.x]);
        atomicAdd(&gsumsq[c], ssq[threadIdx.x]);
    }
}

__global__ __launch_bounds__(256) void final_kernel(
    const float* __restrict__ conv2, const float* __restrict__ x1,
    const float* __restrict__ sum, const float* __restrict__ sumsq,
    const float* __restrict__ gamma, const float* __restrict__ beta,
    float* __restrict__ out) {
    __shared__ float tile[CHN * 33];
    const int p0 = blockIdx.x * 32;
    const int tid = threadIdx.x;
    const float inv = 1.0f / (float)PIX;
    #pragma unroll 4
    for (int i = 0; i < 32; ++i) {
        int d = i * 256 + tid;
        int pl = d >> 8;
        int c = d & 255;
        size_t j = (size_t)(p0 + pl) * CHN + c;
        float mean = sum[c] * inv;
        float var = sumsq[c] * inv - mean * mean;
        float rs = rsqrtf(var + EPSBN);
        tile[c * 33 + pl] = (conv2[j] - mean) * rs * gamma[c] + beta[c] + x1[j];
    }
    __syncthreads();
    #pragma unroll 4
    for (int i = 0; i < 32; ++i) {
        int d = i * 256 + tid;
        int c = d >> 5;
        int pl = d & 31;
        int p = p0 + pl;
        int b = p / (HOUT * HOUT);
        int off = p % (HOUT * HOUT);
        out[((size_t)b * CHN + c) * (HOUT * HOUT) + off] = tile[c * 33 + pl];
    }
}

extern "C" void kernel_launch(void* const* d_in, const int* in_sizes, int n_in,
                              void* d_out, int out_size, void* d_ws, size_t ws_size,
                              hipStream_t stream) {
    const float* x      = (const float*)d_in[0];
    const float* w1     = (const float*)d_in[1];
    const float* w2     = (const float*)d_in[2];
    const float* gamma1 = (const float*)d_in[3];
    const float* beta1  = (const float*)d_in[4];
    const float* gamma2 = (const float*)d_in[5];
    const float* beta2  = (const float*)d_in[6];
    float* out = (float*)d_out;

    (void)in_sizes; (void)n_in; (void)out_size; (void)ws_size;

    char* base = (char*)d_ws;
    size_t off = 0;
    auto take = [&](size_t bytes) -> char* {
        char* ptr = base + off;
        off = (off + bytes + 255) & ~(size_t)255;
        return ptr;
    };
    const size_t xbinBytes = (size_t)BATCH * HIN * HIN * CHN;
    const size_t fmapBytes = (size_t)PIX * CHN * sizeof(float);

    signed char* xbin     = (signed char*)take(xbinBytes);
    signed char* wp1      = (signed char*)take(144 * 1024);
    signed char* wp2      = (signed char*)take(64 * 1024);
    float*       convbuf  = (float*)take(fmapBytes);
    float*       shortcut = (float*)take(fmapBytes);
    float*       x1       = (float*)take(fmapBytes);
    signed char* x1bin    = (signed char*)take((size_t)PIX * CHN);
    float*       stats    = (float*)take(4 * CHN * sizeof(float));
    float* sum1 = stats, *sumsq1 = stats + CHN;
    float* sum2 = stats + 2 * CHN, *sumsq2 = stats + 3 * CHN;

    const int fTotal = PIX * CHN;

    binarize_x_kernel<<<BATCH * HIN, 256, 0, stream>>>(x, xbin);
    maxpool_kernel<<<BATCH * HOUT, 256, 0, stream>>>(x, shortcut);
    pack_w1_kernel<<<36, 128, 0, stream>>>(w1, wp1);
    pack_w2_kernel<<<16, 128, 0, stream>>>(w2, wp2);
    zero_stats_kernel<<<4, 256, 0, stream>>>(stats);
    conv1_wmma_kernel<<<PIX / 4, 128, 0, stream>>>(xbin, wp1, convbuf, sum1, sumsq1);
    bn1_apply_kernel<<<fTotal / 256, 256, 0, stream>>>(
        convbuf, shortcut, sum1, sumsq1, gamma1, beta1, x1, x1bin);
    conv2_wmma_kernel<<<PIX / 4, 128, 0, stream>>>(x1bin, wp2, convbuf, sum2, sumsq2);
    final_kernel<<<PIX / 32, 256, 0, stream>>>(
        convbuf, x1, sum2, sumsq2, gamma2, beta2, out);
}
